// LEFTNet_35150012351074
// MI455X (gfx1250) — matrix operand order repaired and checked by reading, not compiled, not run on hardware
//
#include <hip/hip_runtime.h>
#include <stdint.h>

typedef __bf16 bf16;
typedef __attribute__((ext_vector_type(16))) __bf16 v16bf;
typedef __attribute__((ext_vector_type(8)))  __bf16 bf16x8;
typedef __attribute__((ext_vector_type(4)))  __bf16 bf16x4;
typedef __attribute__((ext_vector_type(8)))  float  v8f;
typedef __attribute__((ext_vector_type(4)))  unsigned tdm4;
typedef __attribute__((ext_vector_type(8)))  unsigned tdm8;

#define INV_SQRT_3F 0.5773502691896258f
#define INV_SQRT_HF 0.08838834764831845f  /* 1/sqrt(128) */

// ---------------------------------------------------------------------------
// WMMA bf16 16x16x32 fragment loaders (wave32).
// ---------------------------------------------------------------------------
__device__ __forceinline__ v16bf load_a_frag(const bf16* base, int stride,
                                             int m0, int k0, int lane) {
  const int h = lane >> 4;
  const int m = m0 + (lane & 15);
  const bf16* p = base + m * stride + k0 + h * 8;
  v16bf a;
  ((bf16x8*)&a)[0] = *(const bf16x8*)(p);
  ((bf16x8*)&a)[1] = *(const bf16x8*)(p + 16);
  return a;
}

__device__ __forceinline__ v16bf load_b_frag(const bf16* base, int stride,
                                             int n0, int lane) {
  const int h = lane >> 4;
  const int n = n0 + (lane & 15);
  const bf16* p = base + n * stride + h * 8;
  v16bf b;
  ((bf16x8*)&b)[0] = *(const bf16x8*)(p);
  ((bf16x8*)&b)[1] = *(const bf16x8*)(p + 16);
  return b;
}

__device__ __forceinline__ bf16x4 cvt4(float4 v) {
  bf16x4 c;
  c[0] = (bf16)v.x; c[1] = (bf16)v.y; c[2] = (bf16)v.z; c[3] = (bf16)v.w;
  return c;
}

__device__ __forceinline__ float silu_f(float v) {
  return v / (1.0f + __expf(-v));
}

// ---------------------------------------------------------------------------
// Tensor Data Mover: one-instruction 2-D tile DMA global -> LDS.
// Tile: `rows` rows x `tile_k` bf16 elems (contiguous), global row stride
// `row_stride` elems. LDS rows padded 64B data + 16B pad -> stride-40 bf16
// (pad_interval code 3 = 16 DWORDs, pad_amount code 3 = 4 DWORDs).
// Tensor dims set huge: tiles are always in-bounds, OOB clamping unused.
// D# per CDNA5 ISA; groups 2/3 omitted (2-D).
// ---------------------------------------------------------------------------
__device__ __forceinline__ void tdm_load_2d(const bf16* g, const bf16* lds,
                                            unsigned tile_k, unsigned rows,
                                            unsigned row_stride) {
  unsigned lo = (unsigned)(uintptr_t)lds;
  unsigned long long ga = (unsigned long long)(uintptr_t)g;
  tdm4 g0;
  g0[0] = 1u;                                   // count=1, user desc
  g0[1] = lo;                                   // lds_addr
  g0[2] = (unsigned)(ga & 0xffffffffu);         // global_addr[31:0]
  g0[3] = (unsigned)((ga >> 32) & 0x01ffffffu)  // global_addr[56:32]
          | (2u << 30);                         // type = 2 ("image")
  tdm8 g1;
  g1[0] = (1u << 16)    // data_size = 2B
        | (1u << 20)    // pad_enable
        | (3u << 22)    // pad_interval: 16 DWORDs (64B)
        | (3u << 25);   // pad_amount: 4 DWORDs (16B)
  g1[1] = 0xffffu << 16;            // tensor_dim0[15:0] (huge)
  g1[2] = 0x7fffu | (0xffffu << 16);  // tensor_dim0[31:16] | tensor_dim1[15:0]
  g1[3] = 0x7fffu | (tile_k << 16);   // tensor_dim1[31:16] | tile_dim0
  g1[4] = rows;                       // tile_dim1 | tile_dim2=0
  g1[5] = row_stride;                 // tensor_dim0_stride[31:0]
  g1[6] = 0u;
  g1[7] = 0u;
  asm volatile("tensor_load_to_lds %0, %1" :: "s"(g0), "s"(g1) : "memory");
}

__device__ __forceinline__ void wait_tensor0() {
#if __has_builtin(__builtin_amdgcn_s_wait_tensorcnt)
  __builtin_amdgcn_s_wait_tensorcnt(0);
#else
  asm volatile("s_wait_tensorcnt 0x0" ::: "memory");
#endif
}

// ---------------------------------------------------------------------------
__global__ void zero_kernel(float* __restrict__ p, int n) {
  int i = blockIdx.x * blockDim.x + threadIdx.x;
  if (i < n) p[i] = 0.0f;
}

// ---------------------------------------------------------------------------
// One-shot weight conversion: fp32 -> bf16, transposed to [n][k] row-major.
// ---------------------------------------------------------------------------
#define PREP_TOTAL (384 * 416 + 384 * 384 + 384 * 32 + 128 * 128 + 384 * 128)

__global__ void prep_weights_kernel(
    const float* __restrict__ dir_w1, const float* __restrict__ dir_w2,
    const float* __restrict__ rbf_w, const float* __restrict__ xp_w1,
    const float* __restrict__ xp_w2, bf16* __restrict__ wB1,
    bf16* __restrict__ wB2, bf16* __restrict__ wBr, bf16* __restrict__ wX1,
    bf16* __restrict__ wX2) {
  int j = blockIdx.x * blockDim.x + threadIdx.x;
  if (j < 384 * 416) {
    wB1[j] = (bf16)dir_w1[(j % 416) * 384 + (j / 416)];
    return;
  }
  j -= 384 * 416;
  if (j < 384 * 384) {
    wB2[j] = (bf16)dir_w2[(j % 384) * 384 + (j / 384)];
    return;
  }
  j -= 384 * 384;
  if (j < 384 * 32) {
    wBr[j] = (bf16)rbf_w[(j % 32) * 384 + (j / 32)];
    return;
  }
  j -= 384 * 32;
  if (j < 128 * 128) {
    wX1[j] = (bf16)xp_w1[(j % 128) * 128 + (j / 128)];
    return;
  }
  j -= 128 * 128;
  if (j < 384 * 128) {
    wX2[j] = (bf16)xp_w2[(j % 128) * 384 + (j / 128)];
  }
}

// ---------------------------------------------------------------------------
// Node projection: xh = silu(LN(x) @ xp_w1) @ xp_w2  -> [N,384] fp32 (d_ws)
// ---------------------------------------------------------------------------
#define BS_STRIDE 40
#define AN_STRIDE 136

__global__ __launch_bounds__(256) void node_xh_kernel(
    const float* __restrict__ x, const float* __restrict__ ln_w,
    const float* __restrict__ ln_b, const bf16* __restrict__ wX1,
    const bf16* __restrict__ wX2, float* __restrict__ xh) {
  __shared__ __align__(16) char usm[384 * BS_STRIDE * 2];  // Bs / Xs union
  __shared__ __align__(16) bf16 An[32 * AN_STRIDE];
  __shared__ __align__(16) bf16 Ht[32 * AN_STRIDE];
  __shared__ float psum[32 * 8];
  __shared__ float mus[32];
  __shared__ float rss[32];
  float* Xs = (float*)usm;
  bf16*  Bs = (bf16*)usm;

  const int tid  = threadIdx.x;
  const int lane = tid & 31;
  const int w    = tid >> 5;
  const int wm   = w & 1;
  const int wn   = w >> 1;
  const int m0   = wm * 16;
  const int row0 = blockIdx.x * 32;

  for (int idx = tid; idx < 32 * 128; idx += 256)
    Xs[idx] = x[(size_t)row0 * 128 + idx];
  __syncthreads();

  {
    int r = tid >> 3, p = tid & 7;
    float s = 0.0f;
    for (int k = p * 16; k < p * 16 + 16; ++k) s += Xs[r * 128 + k];
    psum[r * 8 + p] = s;
  }
  __syncthreads();
  if (tid < 32) {
    float s = 0.0f;
    for (int p = 0; p < 8; ++p) s += psum[tid * 8 + p];
    mus[tid] = s * (1.0f / 128.0f);
  }
  __syncthreads();
  {
    int r = tid >> 3, p = tid & 7;
    float mu = mus[r], s = 0.0f;
    for (int k = p * 16; k < p * 16 + 16; ++k) {
      float d = Xs[r * 128 + k] - mu;
      s += d * d;
    }
    psum[r * 8 + p] = s;
  }
  __syncthreads();
  if (tid < 32) {
    float s = 0.0f;
    for (int p = 0; p < 8; ++p) s += psum[tid * 8 + p];
    rss[tid] = rsqrtf(s * (1.0f / 128.0f) + 1e-5f);
  }
  __syncthreads();

  for (int idx = tid; idx < 32 * 128; idx += 256) {
    int m = idx >> 7, k = idx & 127;
    float v = (Xs[m * 128 + k] - mus[m]) * rss[m] * ln_w[k] + ln_b[k];
    An[m * AN_STRIDE + k] = (bf16)v;
  }
  __syncthreads();  // Xs dead -> Bs

  v8f acc1[2];
#pragma unroll
  for (int t = 0; t < 2; ++t)
#pragma unroll
    for (int r = 0; r < 8; ++r) acc1[t][r] = 0.0f;

  for (int k0 = 0; k0 < 128; k0 += 32) {
    __syncthreads();
    for (int idx = tid; idx < 128 * 4; idx += 256) {
      int n = idx >> 2, c = idx & 3;
      *(bf16x8*)(Bs + n * BS_STRIDE + c * 8) =
          *(const bf16x8*)(wX1 + n * 128 + k0 + c * 8);
    }
    __syncthreads();
    v16bf a = load_a_frag(An, AN_STRIDE, m0, k0, lane);
#pragma unroll
    for (int t = 0; t < 2; ++t) {
      v16bf b = load_b_frag(Bs, BS_STRIDE, wn * 32 + t * 16, lane);
      acc1[t] = __builtin_amdgcn_wmma_f32_16x16x32_bf16(
          false, a, false, b, (short)0, acc1[t], false, false);
    }
  }

  {
    int h = lane >> 4, ln = lane & 15;
#pragma unroll
    for (int t = 0; t < 2; ++t) {
      int n = wn * 32 + t * 16 + ln;
#pragma unroll
      for (int r = 0; r < 8; ++r) {
        int m = m0 + r + 8 * h;
        Ht[m * AN_STRIDE + n] = (bf16)silu_f(acc1[t][r]);
      }
    }
  }
  __syncthreads();

  v8f acc2[6];
#pragma unroll
  for (int t = 0; t < 6; ++t)
#pragma unroll
    for (int r = 0; r < 8; ++r) acc2[t][r] = 0.0f;

  for (int k0 = 0; k0 < 128; k0 += 32) {
    __syncthreads();
    for (int idx = tid; idx < 384 * 4; idx += 256) {
      int n = idx >> 2, c = idx & 3;
      *(bf16x8*)(Bs + n * BS_STRIDE + c * 8) =
          *(const bf16x8*)(wX2 + n * 128 + k0 + c * 8);
    }
    __syncthreads();
    v16bf a = load_a_frag(Ht, AN_STRIDE, m0, k0, lane);
#pragma unroll
    for (int t = 0; t < 6; ++t) {
      v16bf b = load_b_frag(Bs, BS_STRIDE, wn * 96 + t * 16, lane);
      acc2[t] = __builtin_amdgcn_wmma_f32_16x16x32_bf16(
          false, a, false, b, (short)0, acc2[t], false, false);
    }
  }

  {
    int h = lane >> 4, ln = lane & 15;
#pragma unroll
    for (int t = 0; t < 6; ++t) {
      int n = wn * 96 + t * 16 + ln;
#pragma unroll
      for (int r = 0; r < 8; ++r) {
        int m = m0 + r + 8 * h;
        xh[(size_t)(row0 + m) * 384 + n] = acc2[t][r];
      }
    }
  }
}

// ---------------------------------------------------------------------------
// Edge kernel, 64 edges/WG, 8 waves = 4 M-tiles x 2 N-groups.
// B slices arrive via TDM (tensor_load_to_lds, wave 0 issues, TENSORcnt),
// double-buffered; A slices converted fp32->bf16 by all threads.
// Wave (wm,wn) owns n-tiles {part*8 + 2q + wn} -> register-resident epilogue.
// Dynamic LDS: Bs0|Bs1 (2x30720) As0|As1 (2x5120) Arbf (5120) Hs (50176).
// ---------------------------------------------------------------------------
#define HS_STRIDE 392
#define EDGE_LDS_BYTES (30720 * 2 + 5120 * 2 + 5120 + 50176) /* 126976 */

__global__ __launch_bounds__(256) void edge_kernel(
    const float* __restrict__ xh, const float* __restrict__ vec,
    const int* __restrict__ ei, const float* __restrict__ edge_rbf,
    const float* __restrict__ weight, const float* __restrict__ evg,
    const float* __restrict__ ecg, const bf16* __restrict__ wB1,
    const bf16* __restrict__ wB2, const bf16* __restrict__ wBr,
    const float* __restrict__ dir_b1, const float* __restrict__ dir_b2,
    float* __restrict__ dx, float* __restrict__ dvec, int E) {
  extern __shared__ __align__(16) char dsm[];
  bf16* Bs0  = (bf16*)dsm;
  bf16* Bs1  = (bf16*)(dsm + 30720);
  bf16* As0  = (bf16*)(dsm + 61440);
  bf16* As1  = (bf16*)(dsm + 66560);
  bf16* Arbf = (bf16*)(dsm + 71680);
  bf16* Hs   = (bf16*)(dsm + 76800);
  __shared__ int   jjs[64];
  __shared__ int   iis[64];
  __shared__ float evs[192];
  __shared__ float ecs[192];

  const int tid  = threadIdx.x;
  const int lane = tid & 31;
  const int w    = tid >> 5;
  const int wm   = w & 3;
  const int wn   = w >> 2;
  const int m0   = wm * 16;
  const int e0   = blockIdx.x * 64;

  // stage + convert one 64x32 A slice of `weight`
  auto stageA = [&](int k0, bf16* Abuf) {
    for (int idx = tid; idx < 64 * 8; idx += 256) {
      int m = idx >> 3, k4 = (idx & 7) * 4;
      float4 v = *(const float4*)(weight + (size_t)(e0 + m) * 416 + k0 + k4);
      *(bf16x4*)(Abuf + m * 40 + k4) = cvt4(v);
    }
  };

  // Phase 0
  if (tid < 64) {
    jjs[tid] = ei[e0 + tid];
    iis[tid] = ei[E + e0 + tid];
  }
  if (tid < 192) {
    evs[tid] = evg[(size_t)e0 * 3 + tid];
    ecs[tid] = ecg[(size_t)e0 * 3 + tid];
  }
  for (int idx = tid; idx < 64 * 8; idx += 256) {
    int m = idx >> 3, k4 = (idx & 7) * 4;
    float4 v = *(const float4*)(edge_rbf + (size_t)(e0 + m) * 32 + k4);
    *(bf16x4*)(Arbf + m * 40 + k4) = cvt4(v);
  }

  v8f acc[12];
#pragma unroll
  for (int t = 0; t < 12; ++t)
#pragma unroll
    for (int r = 0; r < 8; ++r) acc[t][r] = 0.0f;

  // GEMM1: weight[64,416] @ dir_w1  (13 K-steps, TDM double-buffered)
  if (w == 0) tdm_load_2d(wB1, Bs0, 32, 384, 416);
  stageA(0, As0);
  if (w == 0) wait_tensor0();
  __syncthreads();
  for (int ks = 0; ks < 13; ++ks) {
    bf16* Bcur = (ks & 1) ? Bs1 : Bs0;
    bf16* Acur = (ks & 1) ? As1 : As0;
    if (ks + 1 < 13) {
      if (w == 0)
        tdm_load_2d(wB1 + (ks + 1) * 32, (ks & 1) ? Bs0 : Bs1, 32, 384, 416);
      stageA((ks + 1) * 32, (ks & 1) ? As0 : As1);
    }
    v16bf a = load_a_frag(Acur, 40, m0, 0, lane);
#pragma unroll
    for (int t = 0; t < 12; ++t) {
      int T = (t >> 2) * 8 + (t & 3) * 2 + wn;
      v16bf b = load_b_frag(Bcur, 40, T * 16, lane);
      acc[t] = __builtin_amdgcn_wmma_f32_16x16x32_bf16(
          false, a, false, b, (short)0, acc[t], false, false);
    }
    if (w == 0) wait_tensor0();
    __syncthreads();
  }

  // bias + silu -> Hs
  {
    int h = lane >> 4, ln = lane & 15;
#pragma unroll
    for (int t = 0; t < 12; ++t) {
      int T = (t >> 2) * 8 + (t & 3) * 2 + wn;
      int n = T * 16 + ln;
      float b1 = dir_b1[n];
#pragma unroll
      for (int r = 0; r < 8; ++r) {
        int m = m0 + r + 8 * h;
        Hs[m * HS_STRIDE + n] = (bf16)silu_f(acc[t][r] + b1);
      }
    }
  }

#pragma unroll
  for (int t = 0; t < 12; ++t)
#pragma unroll
    for (int r = 0; r < 8; ++r) acc[t][r] = 0.0f;

  // GEMM2: Hs[64,384] @ dir_w2  (12 K-steps, TDM double-buffered)
  if (w == 0) {
    tdm_load_2d(wB2, Bs0, 32, 384, 384);
    wait_tensor0();
  }
  __syncthreads();  // also publishes Hs
  for (int ks = 0; ks < 12; ++ks) {
    bf16* Bcur = (ks & 1) ? Bs1 : Bs0;
    if (ks + 1 < 12 && w == 0)
      tdm_load_2d(wB2 + (ks + 1) * 32, (ks & 1) ? Bs0 : Bs1, 32, 384, 384);
    v16bf a = load_a_frag(Hs, HS_STRIDE, m0, ks * 32, lane);
#pragma unroll
    for (int t = 0; t < 12; ++t) {
      int T = (t >> 2) * 8 + (t & 3) * 2 + wn;
      v16bf b = load_b_frag(Bcur, 40, T * 16, lane);
      acc[t] = __builtin_amdgcn_wmma_f32_16x16x32_bf16(
          false, a, false, b, (short)0, acc[t], false, false);
    }
    if (w == 0) wait_tensor0();
    __syncthreads();
  }

  // + dir_b2
  {
    int ln = lane & 15;
#pragma unroll
    for (int t = 0; t < 12; ++t) {
      int T = (t >> 2) * 8 + (t & 3) * 2 + wn;
      float b2 = dir_b2[T * 16 + ln];
#pragma unroll
      for (int r = 0; r < 8; ++r) acc[t][r] += b2;
    }
  }

  // rbf B slice (K=32, single TDM stage)
  if (w == 0) {
    tdm_load_2d(wBr, Bs0, 32, 384, 32);
    wait_tensor0();
  }
  __syncthreads();

  // Epilogue: rbf WMMA per column triple, message + scatter from registers
  {
    const int h = lane >> 4, ln = lane & 15;
    v8f zc;
#pragma unroll
    for (int r = 0; r < 8; ++r) zc[r] = 0.0f;
    v16bf arbf = load_a_frag(Arbf, 40, m0, 0, lane);

#pragma unroll
    for (int q = 0; q < 4; ++q) {
      const int T0 = 2 * q + wn;  // x_m tile (cols 0..127)
      v16bf b0 = load_b_frag(Bs0, 40, T0 * 16, lane);
      v16bf b1 = load_b_frag(Bs0, 40, (T0 + 8) * 16, lane);
      v16bf b2 = load_b_frag(Bs0, 40, (T0 + 16) * 16, lane);
      v8f r0 = __builtin_amdgcn_wmma_f32_16x16x32_bf16(
          false, arbf, false, b0, (short)0, zc, false, false);
      v8f r1 = __builtin_amdgcn_wmma_f32_16x16x32_bf16(
          false, arbf, false, b1, (short)0, zc, false, false);
      v8f r2 = __builtin_amdgcn_wmma_f32_16x16x32_bf16(
          false, arbf, false, b2, (short)0, zc, false, false);

      const int hh = T0 * 16 + ln;
#pragma unroll
      for (int r = 0; r < 8; ++r) {
        const int m  = m0 + r + 8 * h;
        const int jj = jjs[m];
        const int ii = iis[m];
        const float* xj = xh + (size_t)jj * 384 + hh;
        const float* xi = xh + (size_t)ii * 384 + hh;
        float xm = (xj[0]   + xi[0])   * r0[r] * acc[q][r];
        float x2 = (xj[128] + xi[128]) * r1[r] * acc[4 + q][r] * INV_SQRT_3F;
        float x3 = (xj[256] + xi[256]) * r2[r] * acc[8 + q][r];
        atomicAdd(&dx[(size_t)ii * 128 + hh], xm);
#pragma unroll
        for (int d = 0; d < 3; ++d) {
          float vj = vec[((size_t)jj * 3 + d) * 128 + hh];
          float vm = (vj * x2 + x3 * evs[m * 3 + d] + xm * ecs[m * 3 + d]) *
                     INV_SQRT_HF;
          atomicAdd(&dvec[((size_t)ii * 3 + d) * 128 + hh], vm);
        }
      }
    }
  }
}

// ---------------------------------------------------------------------------
extern "C" void kernel_launch(void* const* d_in, const int* in_sizes, int n_in,
                              void* d_out, int out_size, void* d_ws,
                              size_t ws_size, hipStream_t stream) {
  const float* x       = (const float*)d_in[0];
  const float* vec     = (const float*)d_in[1];
  const int*   ei      = (const int*)d_in[2];
  const float* e_rbf   = (const float*)d_in[3];
  const float* weight  = (const float*)d_in[4];
  const float* e_vec   = (const float*)d_in[5];
  const float* e_cross = (const float*)d_in[6];
  const float* ln_w    = (const float*)d_in[7];
  const float* ln_b    = (const float*)d_in[8];
  const float* xp_w1   = (const float*)d_in[9];
  const float* xp_w2   = (const float*)d_in[10];
  const float* rbf_w   = (const float*)d_in[11];
  const float* dir_w1  = (const float*)d_in[12];
  const float* dir_b1  = (const float*)d_in[13];
  const float* dir_w2  = (const float*)d_in[14];
  const float* dir_b2  = (const float*)d_in[15];

  const int N = in_sizes[0] / 128;   // 20000
  const int E = in_sizes[3] / 32;    // 320000

  float* out = (float*)d_out;
  char*  ws  = (char*)d_ws;

  float* xhw = (float*)ws;                 // [N,384] f32
  size_t off = (size_t)N * 384 * 4;
  bf16* wB1 = (bf16*)(ws + off); off += 384 * 416 * 2;
  bf16* wB2 = (bf16*)(ws + off); off += 384 * 384 * 2;
  bf16* wBr = (bf16*)(ws + off); off += 384 * 32 * 2;
  bf16* wX1 = (bf16*)(ws + off); off += 128 * 128 * 2;
  bf16* wX2 = (bf16*)(ws + off);

  float* dxp = out;                        // [N,128]
  float* dvp = out + (size_t)N * 128;      // [N,3,128]

  (void)hipFuncSetAttribute((const void*)edge_kernel,
                            hipFuncAttributeMaxDynamicSharedMemorySize,
                            EDGE_LDS_BYTES);

  prep_weights_kernel<<<(PREP_TOTAL + 255) / 256, 256, 0, stream>>>(
      dir_w1, dir_w2, rbf_w, xp_w1, xp_w2, wB1, wB2, wBr, wX1, wX2);
  zero_kernel<<<(out_size + 255) / 256, 256, 0, stream>>>(out, out_size);
  node_xh_kernel<<<N / 32, 256, 0, stream>>>(x, ln_w, ln_b, wX1, wX2, xhw);
  edge_kernel<<<E / 64, 256, EDGE_LDS_BYTES, stream>>>(
      xhw, vec, ei, e_rbf, weight, e_vec, e_cross, wB1, wB2, wBr, dir_b1,
      dir_b2, dxp, dvp, E);
}
